// GNSLayer_54657753809037
// MI455X (gfx1250) — matrix operand order repaired and checked by reading, not compiled, not run on hardware
//
#include <hip/hip_runtime.h>

typedef __attribute__((ext_vector_type(16))) _Float16 v16h;
typedef __attribute__((ext_vector_type(8)))  _Float16 v8h;
typedef __attribute__((ext_vector_type(8)))  float    v8f;

#define HID 64

// ---------------------------------------------------------------- zero ws
__global__ void gns_zero_kernel(float* __restrict__ p, long n) {
  long i = (long)blockIdx.x * blockDim.x + threadIdx.x;
  long stride = (long)gridDim.x * blockDim.x;
  for (; i < n; i += stride) p[i] = 0.0f;
}

// ------------------------------------------------------- scatter-add of e
__global__ void gns_scatter_kernel(const float* __restrict__ e,
                                   const int* __restrict__ ei,
                                   float* __restrict__ agg, int E) {
  long t = (long)blockIdx.x * blockDim.x + threadIdx.x;
  long total = (long)E * HID;
  if (t >= total) return;
  int edge = (int)(t >> 6);
  int f    = (int)(t & 63);
  int r    = ei[edge];
  unsafeAtomicAdd(&agg[(long)r * HID + f], e[t]);
}

// --------------------------------------------------------- edge MLP (WMMA)
// edge_feat = relu([h[row]|h[col]|e] @ eW1 + eb1) @ eW2 + eb2 + e
__global__ __launch_bounds__(256) void gns_edge_kernel(
    const float* __restrict__ h, const float* __restrict__ e,
    const int* __restrict__ ei,
    const float* __restrict__ W1, const float* __restrict__ b1,
    const float* __restrict__ W2, const float* __restrict__ b2,
    float* __restrict__ out, int E) {
  // 24 B-fragments for W1 (K=192) + 8 for W2 (K=64); 512 f16 each = 32 KB
  __shared__ __align__(32) _Float16 wfrag[32 * 512];
  // per-wave 16x64 f16 staging for layer1->layer2 transpose = 16 KB
  __shared__ __align__(32) _Float16 ustage[8 * 16 * HID];

  const int tid = threadIdx.x;
  // Pre-swizzle weights into per-lane WMMA B layout:
  //   lane L holds column n = nt*16 + (L&15); element j -> k = kt*32 + (j>>3)*16 + (L>>4)*8 + (j&7)
  for (int idx = tid; idx < 32 * 512; idx += 256) {
    int f = idx >> 9, w = idx & 511, ln = w >> 4, j = w & 15;
    int n = ln & 15, kh = ln >> 4;
    _Float16 v;
    if (f < 24) {
      int kt = f >> 2, nt = f & 3;
      v = (_Float16)W1[(kt * 32 + (j >> 3) * 16 + kh * 8 + (j & 7)) * HID + nt * 16 + n];
    } else {
      int f2 = f - 24;
      int kt = f2 >> 2, nt = f2 & 3;
      v = (_Float16)W2[(kt * 32 + (j >> 3) * 16 + kh * 8 + (j & 7)) * HID + nt * 16 + n];
    }
    wfrag[idx] = v;
  }
  __syncthreads();

  const int lane = tid & 31;
  const int wave = tid >> 5;
  const int m    = lane & 15;          // A-matrix row within tile
  const int kb   = (lane >> 4) * 8;    // A-fragment k-base
  const int n0   = lane & 15;          // C/D feature within n-tile
  const int mh   = (lane >> 4) << 3;   // C/D row base (0 or 8)
  _Float16* __restrict__ u = &ustage[wave * 16 * HID];

  const int nTiles  = (E + 15) >> 4;
  const int gw0     = blockIdx.x * 8 + wave;
  const int gstride = gridDim.x * 8;

  for (int tile = gw0; tile < nTiles; tile += gstride) {
    const int base = tile << 4;
    int edge = base + m; if (edge >= E) edge = E - 1;
    const int r0 = ei[edge];
    const int c0 = ei[E + edge];
    const float* seg0 = h + (size_t)r0 * HID;
    const float* seg1 = h + (size_t)c0 * HID;
    const float* seg2 = e + (size_t)edge * HID;

    // ---- layer 1: acc = bias, then 6 K-tiles x 4 N-tiles of WMMA ----
    v8f acc[4];
    #pragma unroll
    for (int nt = 0; nt < 4; ++nt) {
      const float bb = b1[nt * 16 + n0];
      #pragma unroll
      for (int r = 0; r < 8; ++r) acc[nt][r] = bb;
    }

    #pragma unroll
    for (int kt = 0; kt < 6; ++kt) {
      const float* p = (kt < 2 ? seg0 : (kt < 4 ? seg1 : seg2)) + (kt & 1) * 32 + kb;
      float4 x0 = *(const float4*)(p + 0);
      float4 x1 = *(const float4*)(p + 4);
      float4 x2 = *(const float4*)(p + 16);
      float4 x3 = *(const float4*)(p + 20);
      v16h a;
      a[0]=(_Float16)x0.x;  a[1]=(_Float16)x0.y;  a[2]=(_Float16)x0.z;  a[3]=(_Float16)x0.w;
      a[4]=(_Float16)x1.x;  a[5]=(_Float16)x1.y;  a[6]=(_Float16)x1.z;  a[7]=(_Float16)x1.w;
      a[8]=(_Float16)x2.x;  a[9]=(_Float16)x2.y;  a[10]=(_Float16)x2.z; a[11]=(_Float16)x2.w;
      a[12]=(_Float16)x3.x; a[13]=(_Float16)x3.y; a[14]=(_Float16)x3.z; a[15]=(_Float16)x3.w;
      #pragma unroll
      for (int nt = 0; nt < 4; ++nt) {
        const v16h bf = *(const v16h*)&wfrag[(kt * 4 + nt) * 512 + lane * 16];
        acc[nt] = __builtin_amdgcn_wmma_f32_16x16x32_f16(
            false, a, false, bf, (short)0, acc[nt], false, false);
      }
    }

    // ---- ReLU + f16 convert, stage 16x64 intermediate in LDS ----
    #pragma unroll
    for (int nt = 0; nt < 4; ++nt) {
      const int n = nt * 16 + n0;
      #pragma unroll
      for (int r = 0; r < 8; ++r) {
        float v = acc[nt][r];
        u[(r + mh) * HID + n] = (_Float16)(v > 0.f ? v : 0.f);
      }
    }

    // ---- layer 2: acc2 = bias + residual e, then 2 K-tiles x 4 N-tiles ----
    v8f acc2[4];
    #pragma unroll
    for (int nt = 0; nt < 4; ++nt) {
      const int n = nt * 16 + n0;
      const float bb = b2[n];
      #pragma unroll
      for (int r = 0; r < 8; ++r) {
        int e2 = base + r + mh; if (e2 >= E) e2 = E - 1;
        acc2[nt][r] = bb + e[(size_t)e2 * HID + n];
      }
    }

    #pragma unroll
    for (int kt = 0; kt < 2; ++kt) {
      const _Float16* up = &u[m * HID + kt * 32 + kb];
      v8h lo = *(const v8h*)(up);
      v8h hi = *(const v8h*)(up + 16);
      v16h a = __builtin_shufflevector(lo, hi, 0,1,2,3,4,5,6,7,8,9,10,11,12,13,14,15);
      #pragma unroll
      for (int nt = 0; nt < 4; ++nt) {
        const v16h bf = *(const v16h*)&wfrag[(24 + kt * 4 + nt) * 512 + lane * 16];
        acc2[nt] = __builtin_amdgcn_wmma_f32_16x16x32_f16(
            false, a, false, bf, (short)0, acc2[nt], false, false);
      }
    }

    // ---- store edge_feat ----
    #pragma unroll
    for (int nt = 0; nt < 4; ++nt) {
      const int n = nt * 16 + n0;
      #pragma unroll
      for (int r = 0; r < 8; ++r) {
        const int e2 = base + r + mh;
        if (e2 < E) out[(size_t)e2 * HID + n] = acc2[nt][r];
      }
    }
  }
}

// --------------------------------------------------------- node MLP (WMMA)
// h_out = relu([h|agg] @ nW1 + nb1) @ nW2 + nb2 + h
__global__ __launch_bounds__(256) void gns_node_kernel(
    const float* __restrict__ h, const float* __restrict__ agg,
    const float* __restrict__ W1, const float* __restrict__ b1,
    const float* __restrict__ W2, const float* __restrict__ b2,
    float* __restrict__ out, int N) {
  __shared__ __align__(32) _Float16 wfrag[24 * 512];   // 16 frags W1 + 8 frags W2
  __shared__ __align__(32) _Float16 ustage[8 * 16 * HID];

  const int tid = threadIdx.x;
  for (int idx = tid; idx < 24 * 512; idx += 256) {
    int f = idx >> 9, w = idx & 511, ln = w >> 4, j = w & 15;
    int n = ln & 15, kh = ln >> 4;
    _Float16 v;
    if (f < 16) {
      int kt = f >> 2, nt = f & 3;
      v = (_Float16)W1[(kt * 32 + (j >> 3) * 16 + kh * 8 + (j & 7)) * HID + nt * 16 + n];
    } else {
      int f2 = f - 16;
      int kt = f2 >> 2, nt = f2 & 3;
      v = (_Float16)W2[(kt * 32 + (j >> 3) * 16 + kh * 8 + (j & 7)) * HID + nt * 16 + n];
    }
    wfrag[idx] = v;
  }
  __syncthreads();

  const int lane = tid & 31;
  const int wave = tid >> 5;
  const int m    = lane & 15;
  const int kb   = (lane >> 4) * 8;
  const int n0   = lane & 15;
  const int mh   = (lane >> 4) << 3;
  _Float16* __restrict__ u = &ustage[wave * 16 * HID];

  const int nTiles  = (N + 15) >> 4;
  const int gw0     = blockIdx.x * 8 + wave;
  const int gstride = gridDim.x * 8;

  for (int tile = gw0; tile < nTiles; tile += gstride) {
    const int base = tile << 4;
    int node = base + m; if (node >= N) node = N - 1;
    const float* seg0 = h   + (size_t)node * HID;
    const float* seg1 = agg + (size_t)node * HID;

    v8f acc[4];
    #pragma unroll
    for (int nt = 0; nt < 4; ++nt) {
      const float bb = b1[nt * 16 + n0];
      #pragma unroll
      for (int r = 0; r < 8; ++r) acc[nt][r] = bb;
    }

    #pragma unroll
    for (int kt = 0; kt < 4; ++kt) {
      const float* p = (kt < 2 ? seg0 : seg1) + (kt & 1) * 32 + kb;
      float4 x0 = *(const float4*)(p + 0);
      float4 x1 = *(const float4*)(p + 4);
      float4 x2 = *(const float4*)(p + 16);
      float4 x3 = *(const float4*)(p + 20);
      v16h a;
      a[0]=(_Float16)x0.x;  a[1]=(_Float16)x0.y;  a[2]=(_Float16)x0.z;  a[3]=(_Float16)x0.w;
      a[4]=(_Float16)x1.x;  a[5]=(_Float16)x1.y;  a[6]=(_Float16)x1.z;  a[7]=(_Float16)x1.w;
      a[8]=(_Float16)x2.x;  a[9]=(_Float16)x2.y;  a[10]=(_Float16)x2.z; a[11]=(_Float16)x2.w;
      a[12]=(_Float16)x3.x; a[13]=(_Float16)x3.y; a[14]=(_Float16)x3.z; a[15]=(_Float16)x3.w;
      #pragma unroll
      for (int nt = 0; nt < 4; ++nt) {
        const v16h bf = *(const v16h*)&wfrag[(kt * 4 + nt) * 512 + lane * 16];
        acc[nt] = __builtin_amdgcn_wmma_f32_16x16x32_f16(
            false, a, false, bf, (short)0, acc[nt], false, false);
      }
    }

    #pragma unroll
    for (int nt = 0; nt < 4; ++nt) {
      const int n = nt * 16 + n0;
      #pragma unroll
      for (int r = 0; r < 8; ++r) {
        float v = acc[nt][r];
        u[(r + mh) * HID + n] = (_Float16)(v > 0.f ? v : 0.f);
      }
    }

    v8f acc2[4];
    #pragma unroll
    for (int nt = 0; nt < 4; ++nt) {
      const int n = nt * 16 + n0;
      const float bb = b2[n];
      #pragma unroll
      for (int r = 0; r < 8; ++r) {
        int n2 = base + r + mh; if (n2 >= N) n2 = N - 1;
        acc2[nt][r] = bb + h[(size_t)n2 * HID + n];   // recurrent residual
      }
    }

    #pragma unroll
    for (int kt = 0; kt < 2; ++kt) {
      const _Float16* up = &u[m * HID + kt * 32 + kb];
      v8h lo = *(const v8h*)(up);
      v8h hi = *(const v8h*)(up + 16);
      v16h a = __builtin_shufflevector(lo, hi, 0,1,2,3,4,5,6,7,8,9,10,11,12,13,14,15);
      #pragma unroll
      for (int nt = 0; nt < 4; ++nt) {
        const v16h bf = *(const v16h*)&wfrag[(16 + kt * 4 + nt) * 512 + lane * 16];
        acc2[nt] = __builtin_amdgcn_wmma_f32_16x16x32_f16(
            false, a, false, bf, (short)0, acc2[nt], false, false);
      }
    }

    #pragma unroll
    for (int nt = 0; nt < 4; ++nt) {
      const int n = nt * 16 + n0;
      #pragma unroll
      for (int r = 0; r < 8; ++r) {
        const int n2 = base + r + mh;
        if (n2 < N) out[(size_t)n2 * HID + n] = acc2[nt][r];
      }
    }
  }
}

// ----------------------------------------------------------------- launch
extern "C" void kernel_launch(void* const* d_in, const int* in_sizes, int n_in,
                              void* d_out, int out_size, void* d_ws, size_t ws_size,
                              hipStream_t stream) {
  const float* h   = (const float*)d_in[0];
  const float* e   = (const float*)d_in[1];
  const int*   ei  = (const int*)  d_in[2];
  const float* eW1 = (const float*)d_in[3];
  const float* eb1 = (const float*)d_in[4];
  const float* eW2 = (const float*)d_in[5];
  const float* eb2 = (const float*)d_in[6];
  // d_in[7..10]: gate MLP — computed but unused in reference outputs, skipped.
  const float* nW1 = (const float*)d_in[11];
  const float* nb1 = (const float*)d_in[12];
  const float* nW2 = (const float*)d_in[13];
  const float* nb2 = (const float*)d_in[14];

  const int N = in_sizes[0] / HID;
  const int E = in_sizes[1] / HID;

  float* h_out = (float*)d_out;
  float* e_out = (float*)d_out + (size_t)N * HID;
  float* agg   = (float*)d_ws;                 // N*HID floats = 25.6 MB scratch

  // edge MLP is independent of the scatter; run it first on the stream
  {
    int tiles  = (E + 15) / 16;
    int blocks = (tiles + 7) / 8;
    gns_edge_kernel<<<blocks, 256, 0, stream>>>(h, e, ei, eW1, eb1, eW2, eb2, e_out, E);
  }

  // zero agg, scatter-add original e by row, then node MLP (stream-ordered)
  gns_zero_kernel<<<1024, 256, 0, stream>>>(agg, (long)N * HID);
  {
    long total = (long)E * HID;
    int blocks = (int)((total + 255) / 256);
    gns_scatter_kernel<<<blocks, 256, 0, stream>>>(e, ei, agg, E);
  }
  {
    int tiles  = (N + 15) / 16;
    int blocks = (tiles + 7) / 8;
    gns_node_kernel<<<blocks, 256, 0, stream>>>(h, agg, nW1, nb1, nW2, nb2, h_out, N);
  }
}
